// TriangleAttentionStartingNode_81862076661847
// MI455X (gfx1250) — matrix operand order repaired and checked by reading, not compiled, not run on hardware
//
#include <hip/hip_runtime.h>
#include <hip/hip_bf16.h>

#define HH 4
#define CC 32
#define CZd 128
#define NNd 256
#define HCd 128

typedef __attribute__((ext_vector_type(16))) __bf16 bf16x16;
typedef __attribute__((ext_vector_type(8)))  float  f32x8;

union Frag16 { bf16x16 v; __bf16 e[16]; uint4 q[2]; };
union FragC  { f32x8 v; float f[8]; };
union U8bf   { uint4 u; __bf16 b[8]; };

__device__ inline f32x8 wmma_bf16(bf16x16 a, bf16x16 b, f32x8 c) {
  return __builtin_amdgcn_wmma_f32_16x16x32_bf16(false, a, false, b, (short)0, c, false, false);
}

__device__ inline uint4 pack8_bf16(const float* f) {
  U8bf p;
#pragma unroll
  for (int i = 0; i < 8; ++i) p.b[i] = (__bf16)f[i];
  return p.u;
}

// fast reciprocal: single v_rcp_f32 (softmax/gate do not need IEEE division)
__device__ inline float fast_rcp(float x) { return __builtin_amdgcn_rcpf(x); }

// ---------------------------------------------------------------------------
// Kernel 0: pre-swizzle the five 128x128 weight matrices into bf16 WMMA
// B-fragments. Dense-B layout: lane l holds column n = nt*16 + (l&15),
// k-dim element e -> k = 32*kc + 8*(l>>4) + (e<8 ? e : 8+e).
// wfrag[mat][kc][nt][lane][e], mats: 0=Wq 1=Wk 2=Wv 3=Wg 4=Wo.
// ---------------------------------------------------------------------------
__global__ void __launch_bounds__(256)
prep_weights_kernel(const float* __restrict__ Wq, const float* __restrict__ Wk,
                    const float* __restrict__ Wv, const float* __restrict__ Wg,
                    const float* __restrict__ Wo, __bf16* __restrict__ wfrag)
{
  int gid = blockIdx.x * 256 + threadIdx.x;
  if (gid >= 5 * 4 * 8 * 32 * 16) return;
  int e   = gid & 15;
  int l   = (gid >> 4) & 31;
  int nt  = (gid >> 9) & 7;
  int kc  = (gid >> 12) & 3;
  int mat = gid >> 14;
  int half = l >> 4, lm = l & 15;
  int kd = 32 * kc + e + 8 * half + ((e >> 3) << 3);
  int n  = nt * 16 + lm;
  const float* W = (mat == 0) ? Wq : (mat == 1) ? Wk : (mat == 2) ? Wv
                 : (mat == 3) ? Wg : Wo;
  wfrag[gid] = (__bf16)W[kd * 128 + n];
}

// ---------------------------------------------------------------------------
// Kernel 1: fused LayerNorm + q/k/v/g/bias projections for a 16-row tile.
// 256 threads = 8 wave32; each wave computes 4 of the 32 (mat, ntile) tiles
// with v_wmma_f32_16x16x32_bf16 over 4 K-chunks.
//   q,k  -> row-major [row][128] bf16
//   v,g  -> transposed [i][h][c][j] bf16  (contiguous B-frag / gate loads)
//   bias -> transposed [h][k][j] f32      (contiguous S-tile bias loads)
// ---------------------------------------------------------------------------
__global__ void __launch_bounds__(256)
ln_proj_kernel(const float* __restrict__ z, const float* __restrict__ lns,
               const float* __restrict__ lnb, const float* __restrict__ Wb,
               const __bf16* __restrict__ wfrag,
               __bf16* __restrict__ q, __bf16* __restrict__ k,
               __bf16* __restrict__ vT, __bf16* __restrict__ gT,
               float* __restrict__ biasT)
{
  __shared__ float red[2][16][16];
  __shared__ float stat[2][16];
  __shared__ __bf16 znb[16][CZd + 8];   // 136 elems = 272 B/row (16B aligned)

  const int t = threadIdx.x;
  const int rbase = blockIdx.x * 16;
  const int row = t >> 4, seg = t & 15;

  float v0[8];
  {
    const float* zp = z + (size_t)(rbase + row) * CZd + seg * 8;
    float s = 0.f, s2 = 0.f;
#pragma unroll
    for (int e2 = 0; e2 < 8; ++e2) { float x = zp[e2]; v0[e2] = x; s += x; s2 += x * x; }
    red[0][row][seg] = s; red[1][row][seg] = s2;
  }
  __syncthreads();
  if (t < 16) {
    float s = 0.f, s2 = 0.f;
    for (int i2 = 0; i2 < 16; ++i2) { s += red[0][t][i2]; s2 += red[1][t][i2]; }
    float mu  = s * (1.f / CZd);
    float var = s2 * (1.f / CZd) - mu * mu;
    stat[0][t] = mu;
    stat[1][t] = rsqrtf(var + 1e-5f);
  }
  __syncthreads();
  {
    float mu = stat[0][row], rs = stat[1][row];
#pragma unroll
    for (int e2 = 0; e2 < 8; ++e2) {
      int c = seg * 8 + e2;
      znb[row][c] = (__bf16)((v0[e2] - mu) * rs * lns[c] + lnb[c]);
    }
  }
  __syncthreads();

  // bias = zn @ Wb  -> biasT[h][k][j]
  if (t < 64) {
    int br = t >> 2, h = t & 3;
    float acc = 0.f;
    for (int c = 0; c < CZd; ++c) acc += (float)znb[br][c] * Wb[c * HH + h];
    int rid = rbase + br;                       // rid = j*256 + kk
    biasT[(size_t)h * 65536 + (size_t)(rid & 255) * 256 + (rid >> 8)] = acc;
  }

  const int lane = t & 31, w = t >> 5;
  const int half = lane >> 4, lm = lane & 15;
  const int ii = rbase >> 8;     // i
  const int jb = rbase & 255;    // j-tile base

#pragma unroll
  for (int it = 0; it < 4; ++it) {
    int tid = w * 4 + it;
    int mat = tid >> 3, nt = tid & 7;
    FragC acc;
#pragma unroll
    for (int r = 0; r < 8; ++r) acc.f[r] = 0.f;
#pragma unroll
    for (int kc = 0; kc < 4; ++kc) {
      Frag16 A, B;
      // A-frag: lane row = lm, k-elems split 8+8 per half (A layout)
      A.q[0] = *(const uint4*)&znb[lm][32 * kc + 8 * half];
      A.q[1] = *(const uint4*)&znb[lm][32 * kc + 16 + 8 * half];
      const __bf16* wp = wfrag + ((((size_t)mat * 4 + kc) * 8 + nt) * 32 + lane) * 16;
      B.q[0] = *(const uint4*)(wp);
      B.q[1] = *(const uint4*)(wp + 8);
      acc.v = wmma_bf16(A.v, B.v, acc.v);
    }
    int col = nt * 16 + lm;
    if (mat <= 1) {
      __bf16* dst = (mat == 0) ? q : k;
      size_t base2 = (size_t)(rbase + 8 * half) * HCd + col;
#pragma unroll
      for (int r = 0; r < 8; ++r) dst[base2 + (size_t)r * HCd] = (__bf16)acc.f[r];
    } else {
      int h = col >> 5, cm = col & 31;
      float vals[8];
      if (mat == 3) {
#pragma unroll
        for (int r = 0; r < 8; ++r)
          vals[r] = fast_rcp(1.f + __expf(-acc.f[r]));   // sigmoid, v_rcp_f32
      } else {
#pragma unroll
        for (int r = 0; r < 8; ++r) vals[r] = acc.f[r];
      }
      __bf16* dst = (mat == 2) ? vT : gT;
      *(uint4*)&dst[(((size_t)ii * HH + h) * CC + cm) * NNd + jb + 8 * half] =
          pack8_bf16(vals);
    }
  }
}

// ---------------------------------------------------------------------------
// Kernel 2: flash attention per (i, h). 8 independent wave32 per workgroup;
// each wave owns two 16-query tiles. Online softmax over 32-key chunks,
// WMMA for QK^T and P·V, per-wave LDS transpose of P (s_wait_dscnt guarded).
// Output gated with gT and written row-major bf16 og[row][128].
// ---------------------------------------------------------------------------
__global__ void __launch_bounds__(256)
attn_kernel(const __bf16* __restrict__ q, const __bf16* __restrict__ k,
            const __bf16* __restrict__ vT, const __bf16* __restrict__ gT,
            const float* __restrict__ biasT, __bf16* __restrict__ og)
{
  __shared__ __bf16 Pt[8][16][40];   // per-wave P transpose tile (80B rows)
  __shared__ float  Osh[8][16][40];  // per-wave output transpose (160B rows)

  const int t = threadIdx.x;
  const int lane = t & 31, w = t >> 5;
  const int half = lane >> 4, lm = lane & 15;
  const int i = blockIdx.x >> 2, h = blockIdx.x & 3;
  const float scale = 0.17677669529663687f;  // 1/sqrt(32)

  const __bf16* kbasep = k  + (size_t)(i * NNd + lm) * HCd + h * CC + 16 * half;
  const __bf16* vbasep = vT + (((size_t)i * HH + h) * CC + lm) * NNd + 16 * half;

  for (int qq = 0; qq < 2; ++qq) {
    const int jb = (w + 8 * qq) * 16;

    Frag16 Qf;  // A-frag of Q
    {
      const __bf16* qp = q + (size_t)(i * NNd + jb + lm) * HCd + h * CC;
      Qf.q[0] = *(const uint4*)(qp + 16 * half);
      Qf.q[1] = *(const uint4*)(qp + 8 + 16 * half);
    }

    FragC O0, O1;
    float M[8], L[8];
#pragma unroll
    for (int r = 0; r < 8; ++r) { O0.f[r] = 0.f; O1.f[r] = 0.f; M[r] = -1e30f; L[r] = 0.f; }

    for (int kb2 = 0; kb2 < 8; ++kb2) {
      // prefetch next 32-key chunk of K and V (global_prefetch_b8) while this
      // chunk's WMMA + softmax executes
      if (kb2 < 7) {
        __builtin_prefetch(kbasep + (size_t)((kb2 + 1) * 32) * HCd, 0, 1);
        __builtin_prefetch(vbasep + (kb2 + 1) * 32, 0, 1);
      }
      FragC S[2];
#pragma unroll
      for (int sub = 0; sub < 2; ++sub) {
        const int kbase = kb2 * 32 + sub * 16;
        Frag16 Kf;  // B-frag of K^T: lane = key column, k-dim = c (contig 16)
        const __bf16* kp = kbasep + (size_t)kbase * HCd;
        Kf.q[0] = *(const uint4*)(kp);
        Kf.q[1] = *(const uint4*)(kp + 8);
        FragC zz;
#pragma unroll
        for (int r = 0; r < 8; ++r) zz.f[r] = 0.f;
        S[sub].v = wmma_bf16(Qf.v, Kf.v, zz.v);
        // bias[j,k,h] in D layout: lane key fixed, 8 consecutive j
        const float* bp = biasT + (size_t)h * 65536 + (size_t)(kbase + lm) * 256 + jb + 8 * half;
        float4 b0 = *(const float4*)(bp);
        float4 b1 = *(const float4*)(bp + 4);
        S[sub].f[0] = S[sub].f[0] * scale + b0.x;
        S[sub].f[1] = S[sub].f[1] * scale + b0.y;
        S[sub].f[2] = S[sub].f[2] * scale + b0.z;
        S[sub].f[3] = S[sub].f[3] * scale + b0.w;
        S[sub].f[4] = S[sub].f[4] * scale + b1.x;
        S[sub].f[5] = S[sub].f[5] * scale + b1.y;
        S[sub].f[6] = S[sub].f[6] * scale + b1.z;
        S[sub].f[7] = S[sub].f[7] * scale + b1.w;
      }
      // online softmax: row reductions across the 16-lane half-groups
      float mt[8], ps[8], corr[8];
#pragma unroll
      for (int r = 0; r < 8; ++r) mt[r] = fmaxf(S[0].f[r], S[1].f[r]);
#pragma unroll
      for (int d = 1; d < 16; d <<= 1)
#pragma unroll
        for (int r = 0; r < 8; ++r) mt[r] = fmaxf(mt[r], __shfl_xor(mt[r], d, 16));
#pragma unroll
      for (int r = 0; r < 8; ++r) {
        float Mn = fmaxf(M[r], mt[r]);
        corr[r] = __expf(M[r] - Mn);
        M[r] = Mn;
        S[0].f[r] = __expf(S[0].f[r] - Mn);
        S[1].f[r] = __expf(S[1].f[r] - Mn);
        ps[r] = S[0].f[r] + S[1].f[r];
        O0.f[r] *= corr[r];
        O1.f[r] *= corr[r];
      }
#pragma unroll
      for (int d = 1; d < 16; d <<= 1)
#pragma unroll
        for (int r = 0; r < 8; ++r) ps[r] += __shfl_xor(ps[r], d, 16);
#pragma unroll
      for (int r = 0; r < 8; ++r) L[r] = L[r] * corr[r] + ps[r];

      // transpose P (D layout -> A layout) through per-wave LDS tile
#pragma unroll
      for (int sub = 0; sub < 2; ++sub)
#pragma unroll
        for (int r = 0; r < 8; ++r)
          Pt[w][r + 8 * half][sub * 16 + lm] = (__bf16)S[sub].f[r];
      asm volatile("s_wait_dscnt 0x0" ::: "memory");

      Frag16 Pf;
      Pf.q[0] = *(const uint4*)&Pt[w][lm][16 * half];
      Pf.q[1] = *(const uint4*)&Pt[w][lm][8 + 16 * half];
#pragma unroll
      for (int ch = 0; ch < 2; ++ch) {
        Frag16 Vf;  // B-frag of V: lane = c column, k-dim = keys (contig 16)
        const __bf16* vp = vbasep + (size_t)(ch * 16) * NNd + kb2 * 32;
        Vf.q[0] = *(const uint4*)(vp);
        Vf.q[1] = *(const uint4*)(vp + 8);
        if (ch == 0) O0.v = wmma_bf16(Pf.v, Vf.v, O0.v);
        else         O1.v = wmma_bf16(Pf.v, Vf.v, O1.v);
      }
      asm volatile("s_wait_dscnt 0x0" ::: "memory");
    }

    // epilogue: normalize, gate, transpose to row-major bf16
    float inv[8];
#pragma unroll
    for (int r = 0; r < 8; ++r) inv[r] = fast_rcp(L[r]);  // v_rcp_f32
#pragma unroll
    for (int ch = 0; ch < 2; ++ch) {
      const int cm = ch * 16 + lm;
      const __bf16* gp = gT + (((size_t)i * HH + h) * CC + cm) * NNd + jb + 8 * half;
      U8bf gv; gv.u = *(const uint4*)gp;
      const FragC& O = ch ? O1 : O0;
#pragma unroll
      for (int r = 0; r < 8; ++r)
        Osh[w][r + 8 * half][cm] = O.f[r] * inv[r] * (float)gv.b[r];
    }
    asm volatile("s_wait_dscnt 0x0" ::: "memory");
    {
      const int orow = lm, c0 = half * 16;
      float4 a0 = *(const float4*)&Osh[w][orow][c0];
      float4 a1 = *(const float4*)&Osh[w][orow][c0 + 4];
      float4 a2 = *(const float4*)&Osh[w][orow][c0 + 8];
      float4 a3 = *(const float4*)&Osh[w][orow][c0 + 12];
      float tv[16] = { a0.x, a0.y, a0.z, a0.w, a1.x, a1.y, a1.z, a1.w,
                       a2.x, a2.y, a2.z, a2.w, a3.x, a3.y, a3.z, a3.w };
      __bf16* op = og + (size_t)(i * NNd + jb + orow) * HCd + h * CC + c0;
      *(uint4*)(op)     = pack8_bf16(tv);
      *(uint4*)(op + 8) = pack8_bf16(tv + 8);
    }
    asm volatile("s_wait_dscnt 0x0" ::: "memory");
  }
}

// ---------------------------------------------------------------------------
// Kernel 3: out = og @ Wo  (fp32 result). 8 waves, one 16x16 output tile each.
// ---------------------------------------------------------------------------
__global__ void __launch_bounds__(256)
outproj_kernel(const __bf16* __restrict__ og, const __bf16* __restrict__ wfrag,
               float* __restrict__ out)
{
  const int t = threadIdx.x;
  const int lane = t & 31, w = t >> 5;
  const int half = lane >> 4, lm = lane & 15;
  const int rbase = blockIdx.x * 16;
  const int nt = w;
  FragC acc;
#pragma unroll
  for (int r = 0; r < 8; ++r) acc.f[r] = 0.f;
#pragma unroll
  for (int kc = 0; kc < 4; ++kc) {
    Frag16 A, B;
    const __bf16* ap = og + (size_t)(rbase + lm) * HCd + 32 * kc;
    A.q[0] = *(const uint4*)(ap + 8 * half);
    A.q[1] = *(const uint4*)(ap + 16 + 8 * half);
    const __bf16* wp = wfrag + ((((size_t)4 * 4 + kc) * 8 + nt) * 32 + lane) * 16;
    B.q[0] = *(const uint4*)(wp);
    B.q[1] = *(const uint4*)(wp + 8);
    acc.v = wmma_bf16(A.v, B.v, acc.v);
  }
  const int col = nt * 16 + lm;
#pragma unroll
  for (int r = 0; r < 8; ++r)
    out[(size_t)(rbase + 8 * half + r) * CZd + col] = acc.f[r];
}

// ---------------------------------------------------------------------------
extern "C" void kernel_launch(void* const* d_in, const int* in_sizes, int n_in,
                              void* d_out, int out_size, void* d_ws, size_t ws_size,
                              hipStream_t stream)
{
  const float* z   = (const float*)d_in[0];
  const float* lns = (const float*)d_in[1];
  const float* lnb = (const float*)d_in[2];
  const float* Wq  = (const float*)d_in[3];
  const float* Wk  = (const float*)d_in[4];
  const float* Wv  = (const float*)d_in[5];
  const float* Wb  = (const float*)d_in[6];
  const float* Wg  = (const float*)d_in[7];
  const float* Wo  = (const float*)d_in[8];
  float* out = (float*)d_out;

  const size_t ROWS = (size_t)NNd * NNd;          // 65536
  char* p = (char*)d_ws;
  __bf16* wfrag = (__bf16*)p;  p += (size_t)5 * 4 * 8 * 32 * 16 * 2;  // 160 KB
  __bf16* q     = (__bf16*)p;  p += ROWS * HCd * 2;                   // 16 MB
  __bf16* k     = (__bf16*)p;  p += ROWS * HCd * 2;                   // 16 MB
  __bf16* vT    = (__bf16*)p;  p += ROWS * HCd * 2;                   // 16 MB
  __bf16* gT    = (__bf16*)p;  p += ROWS * HCd * 2;                   // 16 MB
  float*  biasT = (float*)p;   p += ROWS * HH * 4;                    // 1 MB
  __bf16* og    = (__bf16*)p;  p += ROWS * HCd * 2;                   // 16 MB
  (void)ws_size; (void)in_sizes; (void)n_in; (void)out_size;

  prep_weights_kernel<<<320, 256, 0, stream>>>(Wq, Wk, Wv, Wg, Wo, wfrag);
  ln_proj_kernel<<<4096, 256, 0, stream>>>(z, lns, lnb, Wb, wfrag,
                                           q, k, vT, gT, biasT);
  attn_kernel<<<NNd * HH, 256, 0, stream>>>(q, k, vT, gT, biasT, og);
  outproj_kernel<<<4096, 256, 0, stream>>>(og, wfrag, out);
}
